// UniMPBlock_15109694947895
// MI455X (gfx1250) — compile-verified, matrix operand order checked by reading, not run on hardware
//
#include <hip/hip_runtime.h>
#include <math.h>

// ---------------------------------------------------------------------------
// UniMP / TransformerConv block for MI455X (gfx1250, wave32, WMMA)
// N=50000 nodes, E=100000 edges, D=300, H=4, HC=1200, EDGE_DIM=4
// ---------------------------------------------------------------------------

typedef __attribute__((ext_vector_type(16))) __bf16 v16bf;
typedef __attribute__((ext_vector_type(8)))  __bf16 v8bf;
typedef __attribute__((ext_vector_type(8)))  float  v8f;

#define BMT 128   // block tile M
#define BNT 64    // block tile N
#define BK  32    // k tile (bf16 WMMA depth)
#define LDSK 40   // padded LDS row stride in bf16 (80B, multiple of 16B)

// ---------------------------------------------------------------------------
// Generic GEMM: C[M,Ncols] = A[M,K] * B[K,Ncols] + bias (+residual, +ELU)
// fp32 -> bf16 LDS staging -> V_WMMA_F32_16X16X32_BF16, f32 accum.
// 256 threads = 8 waves; wave (wm,wn) computes a 32x32 output patch.
// Software-pipelined: commit(kt) -> barrier -> load(kt+1) -> compute(kt).
// No masking in the steady state: OOB rows/cols are address-clamped (their
// garbage only lands in never-stored C rows/cols); only the ragged K-tail
// tile is zero-masked, via a peeled uniform branch.
// REQUIRES: K % 4 == 0 (holds: K = 300 or 1200).
// ---------------------------------------------------------------------------
__global__ __launch_bounds__(256)
void wmma_gemm_bf16(const float* __restrict__ A, const float* __restrict__ B,
                    const float* __restrict__ bias, const float* __restrict__ residual,
                    float* __restrict__ C, int M, int Ncols, int K, int do_elu)
{
    __shared__ __bf16 As[2][BMT][LDSK];
    __shared__ __bf16 Bs[2][BNT][LDSK];

    const int t    = threadIdx.x;
    const int lane = t & 31;
    const int wave = t >> 5;
    const int wm   = wave & 3;    // 4 row sub-tiles of 16 (plus +64 sibling)
    const int wn   = wave >> 2;   // 2 col sub-tiles of 32
    const int m0   = blockIdx.y * BMT;
    const int n0   = blockIdx.x * BNT;

    // staging coordinates (constant across k-loop); addresses always in-bounds
    const int  sArow = t >> 1;                               // 0..127
    const int  sAkb  = (t & 1) * 16;                         // 0 or 16
    const float* aBase = A + (size_t)min(m0 + sArow, M - 1) * K;
    const int  sBn   = t & 63;                               // 0..63
    const int  sBkb  = (t >> 6) * 8;                         // 0,8,16,24
    const int  bCol  = min(n0 + sBn, Ncols - 1);

    struct Regs { float4 a[4]; float b[8]; };
    Regs rg;

    // full tile: unmasked clamped loads (garbage only in never-stored rows/cols)
    auto load_full = [&](int k0) {
#pragma unroll
        for (int j = 0; j < 4; ++j)
            rg.a[j] = *(const float4*)(aBase + k0 + sAkb + j * 4);
#pragma unroll
        for (int i = 0; i < 8; ++i)
            rg.b[i] = B[(size_t)(k0 + sBkb + i) * Ncols + bCol];
    };
    // ragged K-tail tile: zero elements with k >= K (one compare per float4 for A)
    auto load_tail = [&](int k0) {
#pragma unroll
        for (int j = 0; j < 4; ++j) {
            const int kk0 = k0 + sAkb + j * 4;
            const float4 f = *(const float4*)(aBase + min(kk0, K - 4));
            const bool ok = (kk0 + 4) <= K;    // K%4==0: all-or-nothing per vec
            rg.a[j].x = ok ? f.x : 0.0f;
            rg.a[j].y = ok ? f.y : 0.0f;
            rg.a[j].z = ok ? f.z : 0.0f;
            rg.a[j].w = ok ? f.w : 0.0f;
        }
#pragma unroll
        for (int i = 0; i < 8; ++i) {
            const int kk = k0 + sBkb + i;
            const float f = B[(size_t)min(kk, K - 1) * Ncols + bCol];
            rg.b[i] = (kk < K) ? f : 0.0f;
        }
    };

    const int  nk       = (K + BK - 1) / BK;
    const bool ragged   = (K % BK) != 0;
    const int  lastFull = ragged ? (nk - 1) : nk;   // tiles [0,lastFull) are full

    auto load_any = [&](int kt) {
        if (kt < lastFull) load_full(kt * BK);
        else               load_tail(kt * BK);
    };

    // convert staged registers to bf16 and commit to LDS buffer `buf`
    auto commit = [&](int buf) {
        v8bf h0, h1, hb;
#pragma unroll
        for (int j = 0; j < 2; ++j) {
            h0[j * 4 + 0] = (__bf16)rg.a[j].x;     h0[j * 4 + 1] = (__bf16)rg.a[j].y;
            h0[j * 4 + 2] = (__bf16)rg.a[j].z;     h0[j * 4 + 3] = (__bf16)rg.a[j].w;
            h1[j * 4 + 0] = (__bf16)rg.a[j + 2].x; h1[j * 4 + 1] = (__bf16)rg.a[j + 2].y;
            h1[j * 4 + 2] = (__bf16)rg.a[j + 2].z; h1[j * 4 + 3] = (__bf16)rg.a[j + 2].w;
        }
#pragma unroll
        for (int i = 0; i < 8; ++i) hb[i] = (__bf16)rg.b[i];
        *(v8bf*)&As[buf][sArow][sAkb]     = h0;
        *(v8bf*)&As[buf][sArow][sAkb + 8] = h1;
        *(v8bf*)&Bs[buf][sBn][sBkb]       = hb;
    };

    v8f acc00 = {}, acc01 = {}, acc10 = {}, acc11 = {};

    const int half = (lane < 16) ? 0 : 8;
    const int mrow = wm * 16 + (lane & 15);
    const int nrow = wn * 32 + (lane & 15);

    load_any(0);
    for (int kt = 0; kt < nk; ++kt) {
        const int buf = kt & 1;
        commit(buf);            // waits for loads issued last iteration
        __syncthreads();        // publish buf; prior readers of buf long done
        if (kt + 1 < nk) load_any(kt + 1);   // overlaps with WMMAs below

        v16bf a0, a1, b0, b1;
#pragma unroll
        for (int i = 0; i < 8; ++i) {
            a0[i]     = As[buf][mrow][half + i];
            a0[i + 8] = As[buf][mrow][half + 16 + i];
            a1[i]     = As[buf][64 + mrow][half + i];
            a1[i + 8] = As[buf][64 + mrow][half + 16 + i];
            b0[i]     = Bs[buf][nrow][half + i];
            b0[i + 8] = Bs[buf][nrow][half + 16 + i];
            b1[i]     = Bs[buf][nrow + 16][half + i];
            b1[i + 8] = Bs[buf][nrow + 16][half + 16 + i];
        }
        acc00 = __builtin_amdgcn_wmma_f32_16x16x32_bf16(false, a0, false, b0, (short)0, acc00, false, false);
        acc01 = __builtin_amdgcn_wmma_f32_16x16x32_bf16(false, a0, false, b1, (short)0, acc01, false, false);
        acc10 = __builtin_amdgcn_wmma_f32_16x16x32_bf16(false, a1, false, b0, (short)0, acc10, false, false);
        acc11 = __builtin_amdgcn_wmma_f32_16x16x32_bf16(false, a1, false, b1, (short)0, acc11, false, false);
    }

    // ---- epilogue (C layout: VGPR r -> row r (+8 upper lanes), N = lane&15)
    const int rbase = (lane < 16) ? 0 : 8;
    const int cn    = lane & 15;
    const int gn0   = n0 + wn * 32 + cn;
    const int gn1   = gn0 + 16;
    const float bv0 = (gn0 < Ncols) ? bias[gn0] : 0.0f;
    const float bv1 = (gn1 < Ncols) ? bias[gn1] : 0.0f;

    auto emit = [&](float accv, int gm, int gn, float bv) {
        if (gm >= M || gn >= Ncols) return;
        float v = accv + bv;
        if (residual) v += residual[(size_t)gm * Ncols + gn];
        if (do_elu)   v = (v > 0.0f) ? v : expm1f(v);
        C[(size_t)gm * Ncols + gn] = v;
    };
#pragma unroll
    for (int r = 0; r < 8; ++r) {
        const int gmA = m0 + wm * 16 + rbase + r;
        const int gmB = gmA + 64;
        emit(acc00[r], gmA, gn0, bv0);
        emit(acc01[r], gmA, gn1, bv1);
        emit(acc10[r], gmB, gn0, bv0);
        emit(acc11[r], gmB, gn1, bv1);
    }
}

// ---------------------------------------------------------------------------
// Float segment-max via monotonic uint encoding
// ---------------------------------------------------------------------------
__device__ __forceinline__ unsigned enc_f32(float f) {
    unsigned u = __float_as_uint(f);
    return (u & 0x80000000u) ? ~u : (u | 0x80000000u);
}
__device__ __forceinline__ float dec_f32(unsigned key) {
    unsigned u = (key & 0x80000000u) ? (key & 0x7FFFFFFFu) : ~key;
    return __uint_as_float(u);
}

__global__ void zero_u32_kernel(unsigned* __restrict__ p, int n) {
    int i = blockIdx.x * blockDim.x + threadIdx.x;
    if (i < n) p[i] = 0u;
}

// ---------------------------------------------------------------------------
// Edge attention logits: one wave per edge. alpha_raw = scale * q[dst].(k[src]+ee)
// ---------------------------------------------------------------------------
__global__ __launch_bounds__(256)
void edge_attn_kernel(const float* __restrict__ q, const float* __restrict__ k,
                      const float* __restrict__ e_attr, const float* __restrict__ We,
                      const float* __restrict__ be, const int* __restrict__ eidx,
                      float* __restrict__ alpha, unsigned* __restrict__ amaxu, int E)
{
    const int gt   = blockIdx.x * blockDim.x + threadIdx.x;
    const int e    = gt >> 5;
    const int lane = threadIdx.x & 31;
    if (e >= E) return;
    const int src = eidx[e];
    const int dst = eidx[E + e];
    const float ea0 = e_attr[e * 4 + 0], ea1 = e_attr[e * 4 + 1];
    const float ea2 = e_attr[e * 4 + 2], ea3 = e_attr[e * 4 + 3];
    const float scale = 0.05773502691896258f;  // 1/sqrt(300)

#pragma unroll
    for (int h = 0; h < 4; ++h) {
        float partial = 0.0f;
        for (int c = lane; c < 300; c += 32) {
            const int col = h * 300 + c;
            const float ee = be[col] + ea0 * We[col] + ea1 * We[1200 + col]
                           + ea2 * We[2400 + col] + ea3 * We[3600 + col];
            partial += q[(size_t)dst * 1200 + col] * (k[(size_t)src * 1200 + col] + ee);
        }
#pragma unroll
        for (int off = 16; off > 0; off >>= 1) partial += __shfl_xor(partial, off, 32);
        if (lane == 0) {
            const float a = partial * scale;
            alpha[e * 4 + h] = a;
            atomicMax(&amaxu[dst * 4 + h], enc_f32(a));
        }
    }
}

// ---------------------------------------------------------------------------
// Softmax numerator + segment denominator: one thread per (edge, head)
// ---------------------------------------------------------------------------
__global__ void edge_softmax_kernel(const int* __restrict__ eidx,
                                    const unsigned* __restrict__ amaxu,
                                    float* __restrict__ alpha,
                                    float* __restrict__ denom, int E)
{
    const int i = blockIdx.x * blockDim.x + threadIdx.x;
    if (i >= E * 4) return;
    const int e = i >> 2;
    const int h = i & 3;
    const int dst = eidx[E + e];
    const float amax = dec_f32(amaxu[dst * 4 + h]);
    const float ex = expf(alpha[i] - amax);
    alpha[i] = ex;
    atomicAdd(&denom[dst * 4 + h], ex);
}

// ---------------------------------------------------------------------------
// Weighted message scatter: one wave per edge. x[dst] += (v[src]+ee)*alpha
// ---------------------------------------------------------------------------
__global__ __launch_bounds__(256)
void edge_message_kernel(const float* __restrict__ v, const float* __restrict__ e_attr,
                         const float* __restrict__ We, const float* __restrict__ be,
                         const int* __restrict__ eidx, const float* __restrict__ alpha,
                         const float* __restrict__ denom, float* __restrict__ x, int E)
{
    const int gt   = blockIdx.x * blockDim.x + threadIdx.x;
    const int e    = gt >> 5;
    const int lane = threadIdx.x & 31;
    if (e >= E) return;
    const int src = eidx[e];
    const int dst = eidx[E + e];
    const float ea0 = e_attr[e * 4 + 0], ea1 = e_attr[e * 4 + 1];
    const float ea2 = e_attr[e * 4 + 2], ea3 = e_attr[e * 4 + 3];

#pragma unroll
    for (int h = 0; h < 4; ++h) {
        const float coef = alpha[e * 4 + h] / (denom[dst * 4 + h] + 1e-16f);
        for (int c = lane; c < 300; c += 32) {
            const int col = h * 300 + c;
            const float ee = be[col] + ea0 * We[col] + ea1 * We[1200 + col]
                           + ea2 * We[2400 + col] + ea3 * We[3600 + col];
            const float msg = (v[(size_t)src * 1200 + col] + ee) * coef;
            atomicAdd(&x[(size_t)dst * 1200 + col], msg);
        }
    }
}

// ---------------------------------------------------------------------------
// LayerNorm over HC=1200 per node, one wave per node, in place.
// ---------------------------------------------------------------------------
__global__ __launch_bounds__(256)
void layernorm_kernel(float* __restrict__ x, const float* __restrict__ ln_a,
                      const float* __restrict__ ln_b, int Nn)
{
    const int gt   = blockIdx.x * blockDim.x + threadIdx.x;
    const int node = gt >> 5;
    const int lane = threadIdx.x & 31;
    if (node >= Nn) return;
    float s = 0.0f, s2 = 0.0f;
    for (int c = lane; c < 1200; c += 32) {
        const float v = x[(size_t)node * 1200 + c];
        s += v; s2 += v * v;
    }
#pragma unroll
    for (int off = 16; off > 0; off >>= 1) {
        s  += __shfl_xor(s,  off, 32);
        s2 += __shfl_xor(s2, off, 32);
    }
    const float mean = s * (1.0f / 1200.0f);
    const float var  = s2 * (1.0f / 1200.0f) - mean * mean;
    const float rstd = rsqrtf(var + 1e-5f);
    for (int c = lane; c < 1200; c += 32) {
        const size_t idx = (size_t)node * 1200 + c;
        x[idx] = ln_a[c] * (x[idx] - mean) * rstd + ln_b[c];
    }
}

// ---------------------------------------------------------------------------
// Launch
// ---------------------------------------------------------------------------
extern "C" void kernel_launch(void* const* d_in, const int* in_sizes, int n_in,
                              void* d_out, int out_size, void* d_ws, size_t ws_size,
                              hipStream_t stream) {
    const int Dd = 300, Hh = 4, HC = 1200;

    const float* xin    = (const float*)d_in[0];
    const int*   eidx   = (const int*)  d_in[1];
    const float* e_attr = (const float*)d_in[2];
    const float* Wq     = (const float*)d_in[3];
    const float* bq     = (const float*)d_in[4];
    const float* Wk     = (const float*)d_in[5];
    const float* bk     = (const float*)d_in[6];
    const float* Wv     = (const float*)d_in[7];
    const float* bv     = (const float*)d_in[8];
    const float* We     = (const float*)d_in[9];
    const float* be     = (const float*)d_in[10];
    const float* Wskip  = (const float*)d_in[11];
    const float* bskip  = (const float*)d_in[12];
    const float* ln_a   = (const float*)d_in[13];
    const float* ln_b   = (const float*)d_in[14];
    const float* Wlin   = (const float*)d_in[15];
    const float* blin   = (const float*)d_in[16];

    const int Nn = in_sizes[0] / Dd;
    const int E  = in_sizes[1] / 2;
    float* out = (float*)d_out;

    // workspace layout
    const size_t NHC = (size_t)Nn * HC;
    float*    qb    = (float*)d_ws;
    float*    kb    = qb + NHC;
    float*    vb    = kb + NHC;
    float*    xb    = vb + NHC;
    float*    alpha = xb + NHC;                            // E*H
    unsigned* amaxu = (unsigned*)(alpha + (size_t)E * Hh); // N*H (then denom N*H)
    float*    denom = (float*)(amaxu + (size_t)Nn * Hh);

    const dim3 blk(256);

    // q/k/v/skip projections (WMMA bf16 GEMMs)
    dim3 g1((HC + BNT - 1) / BNT, (Nn + BMT - 1) / BMT);
    wmma_gemm_bf16<<<g1, blk, 0, stream>>>(xin, Wq,    bq,    nullptr, qb, Nn, HC, Dd, 0);
    wmma_gemm_bf16<<<g1, blk, 0, stream>>>(xin, Wk,    bk,    nullptr, kb, Nn, HC, Dd, 0);
    wmma_gemm_bf16<<<g1, blk, 0, stream>>>(xin, Wv,    bv,    nullptr, vb, Nn, HC, Dd, 0);
    wmma_gemm_bf16<<<g1, blk, 0, stream>>>(xin, Wskip, bskip, nullptr, xb, Nn, HC, Dd, 0);

    // zero segment-max keys and denominators (contiguous: amaxu then denom)
    const int zn = 2 * Nn * Hh;
    zero_u32_kernel<<<(zn + 255) / 256, blk, 0, stream>>>(amaxu, zn);

    // edge attention -> segment softmax -> message scatter
    const int ethreads = E * 32;
    edge_attn_kernel<<<(ethreads + 255) / 256, blk, 0, stream>>>(qb, kb, e_attr, We, be,
                                                                 eidx, alpha, amaxu, E);
    edge_softmax_kernel<<<(E * Hh + 255) / 256, blk, 0, stream>>>(eidx, amaxu, alpha, denom, E);
    edge_message_kernel<<<(ethreads + 255) / 256, blk, 0, stream>>>(vb, e_attr, We, be, eidx,
                                                                    alpha, denom, xb, E);

    // LayerNorm (in place on xb)
    layernorm_kernel<<<((Nn * 32) + 255) / 256, blk, 0, stream>>>(xb, ln_a, ln_b, Nn);

    // output linear + residual + ELU (WMMA bf16 GEMM, fused epilogue)
    dim3 g2((Dd + BNT - 1) / BNT, (Nn + BMT - 1) / BMT);
    wmma_gemm_bf16<<<g2, blk, 0, stream>>>(xb, Wlin, blin, xin, out, Nn, Dd, HC, 1);
}
